// VectorQuantizer_30039001268585
// MI455X (gfx1250) — matrix operand order repaired and checked by reading, not compiled, not run on hardware
//
#include <hip/hip_runtime.h>
#include <hip/hip_bf16.h>

typedef __attribute__((ext_vector_type(16))) __bf16 v16bf;
typedef __attribute__((ext_vector_type(8)))  float  v8f;

#define N_TOK   8192
#define DIM     256
#define K_CB    8192
#define NTOT    2097152      // 8*256*32*32
#define BETA    0.25f
#define EPS_F   1e-10f
#define NSPLIT  8            // column splits
#define JTILES  512          // K_CB/16 column tiles

// d_out layout (floats, concatenated in reference return order)
#define OFF_OUT     0
#define OFF_LOSS    2097152
#define OFF_PERP    2097153
#define OFF_ONEHOT  2097154
#define OFF_IDX     69206018
#define ONEHOT_LEN  67108864

// ---------- bf16 helpers (RNE, bit-exact) ----------
__device__ __forceinline__ unsigned short f2bf(float f) {
    unsigned int u = __float_as_uint(f);
    unsigned int r = u + 0x7FFFu + ((u >> 16) & 1u);
    return (unsigned short)(r >> 16);
}
__device__ __forceinline__ float bf2f(unsigned short s) {
    return __uint_as_float(((unsigned int)s) << 16);
}

// ---------- init small ws accumulators ----------
__global__ void vq_init(int* counts, float* accum) {
    int i = blockIdx.x * blockDim.x + threadIdx.x;
    if (i < K_CB) counts[i] = 0;
    if (i < 2)    accum[i] = 0.0f;
}

// ---------- zero-fill the one-hot region ----------
__global__ void vq_fill_zero(float2* p, long long n2) {
    long long stride = (long long)gridDim.x * blockDim.x;
    for (long long i = blockIdx.x * (long long)blockDim.x + threadIdx.x; i < n2; i += stride)
        p[i] = make_float2(0.0f, 0.0f);
}

// ---------- transpose x -> token-major, split to bf16 hi/lo, fp32 row norms ----------
__global__ void vq_prep_x(const float* __restrict__ x, unsigned short* __restrict__ xh,
                          unsigned short* __restrict__ xl, float* __restrict__ xnorm) {
    int n = blockIdx.x, c = threadIdx.x;
    int b = n >> 10, rem = n & 1023, h = rem >> 5, w = rem & 31;
    float v = x[(b * DIM + c) * 1024 + (h << 5) + w];
    unsigned short hi = f2bf(v);
    unsigned short lo = f2bf(v - bf2f(hi));
    xh[n * DIM + c] = hi;
    xl[n * DIM + c] = lo;
    float s = v * v;
    for (int m = 16; m; m >>= 1) s += __shfl_xor(s, m, 32);
    __shared__ float sm[8];
    if ((threadIdx.x & 31) == 0) sm[threadIdx.x >> 5] = s;
    __syncthreads();
    if (threadIdx.x == 0) {
        float t = 0.0f;
        for (int i = 0; i < 8; ++i) t += sm[i];
        xnorm[n] = t;
    }
}

// ---------- codebook split + norms ----------
__global__ void vq_prep_cb(const float* __restrict__ cb, unsigned short* __restrict__ ch,
                           unsigned short* __restrict__ cl, float* __restrict__ cnorm) {
    int k = blockIdx.x, c = threadIdx.x;
    float v = cb[k * DIM + c];
    unsigned short hi = f2bf(v);
    unsigned short lo = f2bf(v - bf2f(hi));
    ch[k * DIM + c] = hi;
    cl[k * DIM + c] = lo;
    float s = v * v;
    for (int m = 16; m; m >>= 1) s += __shfl_xor(s, m, 32);
    __shared__ float sm[8];
    if ((threadIdx.x & 31) == 0) sm[threadIdx.x >> 5] = s;
    __syncthreads();
    if (threadIdx.x == 0) {
        float t = 0.0f;
        for (int i = 0; i < 8; ++i) t += sm[i];
        cnorm[k] = t;
    }
}

// ---------- fused split-bf16 WMMA distance GEMM + per-row argmin ----------
// grid: (64 row-blocks of 128 rows, 8 column splits of 64 tiles). 8 waves/block,
// one 16-row tile per wave (A in VGPRs), B tiles double-buffered in LDS via
// global_load_async_to_lds_b128 (ASYNCcnt) shared by all 8 waves.
__global__ __launch_bounds__(256) void vq_dist_argmin(
    const unsigned short* __restrict__ xh, const unsigned short* __restrict__ xl,
    const unsigned short* __restrict__ ch, const unsigned short* __restrict__ cl,
    const float* __restrict__ xnorm, const float* __restrict__ cnorm,
    float* __restrict__ pmin, int* __restrict__ pidx) {
    const int lane = threadIdx.x & 31;
    const int wave = threadIdx.x >> 5;
    const int hlf  = lane >> 4;        // 0: lanes 0-15, 1: lanes 16-31
    const int l16  = lane & 15;
    const int rw0  = blockIdx.x * 128 + wave * 16;   // this wave's 16 rows
    const int s    = blockIdx.y;                     // column split
    const int jbeg = s * (JTILES / NSPLIT);
    const int jend = jbeg + (JTILES / NSPLIT);

    __shared__ char  lds_b[2][16384];   // [buf][ hi 8KB | lo 8KB ]
    __shared__ float lds_cn[K_CB];      // codebook norms (32KB)

    for (int k = threadIdx.x; k < K_CB; k += 256) lds_cn[k] = cnorm[k];

    union AF { uint4 u[2]; v16bf v; };
    // 16-bit A layout: lanes 0-15 -> M=lane, K {0..7,16..23}; lanes 16-31 -> K {8..15,24..31}
    v16bf ah[8], al[8];
    {
        const int row = rw0 + l16;
        const int kb  = hlf ? 8 : 0;
        const unsigned short* ph = xh + row * DIM;
        const unsigned short* pl = xl + row * DIM;
#pragma unroll
        for (int t = 0; t < 8; ++t) {
            AF f;
            f.u[0] = *(const uint4*)(ph + 32 * t + kb);
            f.u[1] = *(const uint4*)(ph + 32 * t + kb + 16);
            ah[t] = f.v;
            f.u[0] = *(const uint4*)(pl + 32 * t + kb);
            f.u[1] = *(const uint4*)(pl + 32 * t + kb + 16);
            al[t] = f.v;
        }
    }
    // C/D layout: VGPR r, lanes 0-15 -> M=r; lanes 16-31 -> M=8+r; N = lane%16
    float xn[8];
    {
        const int rb = rw0 + hlf * 8;
#pragma unroll
        for (int r = 0; r < 8; ++r) xn[r] = xnorm[rb + r];
    }
    float rmin[8]; int ridx[8];
#pragma unroll
    for (int r = 0; r < 8; ++r) { rmin[r] = 3.4e38f; ridx[r] = 0; }

    const unsigned ldsb = (unsigned)(size_t)&lds_b[0][0];  // flat addr low 32b == LDS offset

    auto issue = [&](int jj, int b) {
        // tile jj: 16 codebook rows -> 8KB hi + 8KB lo, contiguous in global
        const unsigned long long gh = (unsigned long long)(size_t)((const char*)ch + (size_t)jj * 8192);
        const unsigned long long gl = (unsigned long long)(size_t)((const char*)cl + (size_t)jj * 8192);
        const unsigned base = ldsb + (unsigned)b * 16384u;
        const unsigned o1 = (unsigned)threadIdx.x * 16u;
        const unsigned o2 = o1 + 4096u;
        asm volatile("global_load_async_to_lds_b128 %0, %1, off" :: "v"(base + o1),          "v"(gh + o1) : "memory");
        asm volatile("global_load_async_to_lds_b128 %0, %1, off" :: "v"(base + o2),          "v"(gh + o2) : "memory");
        asm volatile("global_load_async_to_lds_b128 %0, %1, off" :: "v"(base + 8192u + o1),  "v"(gl + o1) : "memory");
        asm volatile("global_load_async_to_lds_b128 %0, %1, off" :: "v"(base + 8192u + o2),  "v"(gl + o2) : "memory");
    };

    issue(jbeg, 0);   // prologue: fill buffer 0

    for (int j = jbeg; j < jend; ++j) {
        asm volatile("s_wait_asynccnt 0x0" ::: "memory");  // tile j resident in LDS
        __syncthreads();  // publishes tile j; also retires all readers of the buffer we refill next
        const int b = (j - jbeg) & 1;
        if (j + 1 < jend) issue(j + 1, b ^ 1);

        const char* lb = &lds_b[b][0];
        const int boff = l16 * 512 + hlf * 16;   // col l16, K {8h..8h+7, 16+8h..16+8h+7}
        v8f acc = {};
#pragma unroll
        for (int t = 0; t < 8; ++t) {
            AF fb;
            fb.u[0] = *(const uint4*)(const void*)(lb + boff + 64 * t);
            fb.u[1] = *(const uint4*)(const void*)(lb + boff + 64 * t + 32);
            v16bf bh = fb.v;
            fb.u[0] = *(const uint4*)(const void*)(lb + 8192 + boff + 64 * t);
            fb.u[1] = *(const uint4*)(const void*)(lb + 8192 + boff + 64 * t + 32);
            v16bf bl = fb.v;
            acc = __builtin_amdgcn_wmma_f32_16x16x32_bf16(false, ah[t], false, bh, (short)0, acc, false, false);
            acc = __builtin_amdgcn_wmma_f32_16x16x32_bf16(false, ah[t], false, bl, (short)0, acc, false, false);
            acc = __builtin_amdgcn_wmma_f32_16x16x32_bf16(false, al[t], false, bh, (short)0, acc, false, false);
        }
        const int col = j * 16 + l16;
        const float cn = lds_cn[col];
#pragma unroll
        for (int r = 0; r < 8; ++r) {
            float d = xn[r] + cn - 2.0f * acc[r];
            if (d < rmin[r] || (d == rmin[r] && col < ridx[r])) { rmin[r] = d; ridx[r] = col; }
        }
    }
    // reduce across the 16 lanes within each half-group (xor masks stay inside the group)
#pragma unroll
    for (int m = 1; m < 16; m <<= 1) {
#pragma unroll
        for (int r = 0; r < 8; ++r) {
            float om = __shfl_xor(rmin[r], m, 32);
            int   oi = __shfl_xor(ridx[r], m, 32);
            if (om < rmin[r] || (om == rmin[r] && oi < ridx[r])) { rmin[r] = om; ridx[r] = oi; }
        }
    }
    if (l16 == 0) {
        const int rb = s * N_TOK + rw0 + hlf * 8;
#pragma unroll
        for (int r = 0; r < 8; ++r) { pmin[rb + r] = rmin[r]; pidx[rb + r] = ridx[r]; }
    }
}

// ---------- combine column-split partial argmins ----------
__global__ void vq_amin_combine(const float* __restrict__ pmin, const int* __restrict__ pidx,
                                int* __restrict__ amin) {
    int row = blockIdx.x * blockDim.x + threadIdx.x;
    float bm = pmin[row]; int bi = pidx[row];
    for (int s = 1; s < NSPLIT; ++s) {
        float om = pmin[s * N_TOK + row]; int oi = pidx[s * N_TOK + row];
        if (om < bm || (om == bm && oi < bi)) { bm = om; bi = oi; }
    }
    amin[row] = bi;
}

// ---------- epilogue: quantized output, one-hot ones, indices, loss sums, histogram ----------
__global__ void vq_scatter(const float* __restrict__ x, const float* __restrict__ cb,
                           const int* __restrict__ amin, float* __restrict__ out,
                           int* counts, float* accum) {
    int n = blockIdx.x, c = threadIdx.x;
    int b = n >> 10, rem = n & 1023, h = rem >> 5, w = rem & 31;
    int idx = amin[n];
    float xq = cb[idx * DIM + c];
    int oi = (b * DIM + c) * 1024 + (h << 5) + w;
    out[OFF_OUT + oi] = xq;                       // straight-through forward == x_q
    float d = x[oi] - xq;
    float s1 = d, s2 = d * d;
    for (int m = 16; m; m >>= 1) { s1 += __shfl_xor(s1, m, 32); s2 += __shfl_xor(s2, m, 32); }
    __shared__ float sm1[8], sm2[8];
    if ((threadIdx.x & 31) == 0) { sm1[threadIdx.x >> 5] = s1; sm2[threadIdx.x >> 5] = s2; }
    __syncthreads();
    if (threadIdx.x == 0) {
        float t1 = 0.0f, t2 = 0.0f;
        for (int i = 0; i < 8; ++i) { t1 += sm1[i]; t2 += sm2[i]; }
        atomicAdd(&accum[0], t1);
        atomicAdd(&accum[1], t2);
        out[OFF_ONEHOT + (long long)n * K_CB + idx] = 1.0f;
        out[OFF_IDX + n] = (float)idx;
        atomicAdd(&counts[idx], 1);
    }
}

// ---------- perplexity + loss scalars ----------
__global__ void vq_finalize(const int* __restrict__ counts, const float* __restrict__ accum,
                            float* __restrict__ out) {
    float s = 0.0f;
    for (int k = threadIdx.x; k < K_CB; k += 256) {
        float e = (float)counts[k] * (1.0f / (float)N_TOK);
        s += e * logf(e + EPS_F);
    }
    for (int m = 16; m; m >>= 1) s += __shfl_xor(s, m, 32);
    __shared__ float sm[8];
    if ((threadIdx.x & 31) == 0) sm[threadIdx.x >> 5] = s;
    __syncthreads();
    if (threadIdx.x == 0) {
        float t = 0.0f;
        for (int i = 0; i < 8; ++i) t += sm[i];
        out[OFF_PERP] = expf(-t);
        out[OFF_LOSS] = BETA * (accum[0] / (float)NTOT) + accum[1] / (float)NTOT;
    }
}

extern "C" void kernel_launch(void* const* d_in, const int* in_sizes, int n_in,
                              void* d_out, int out_size, void* d_ws, size_t ws_size,
                              hipStream_t stream) {
    const float* x  = (const float*)d_in[0];
    const float* cb = (const float*)d_in[1];
    float* out = (float*)d_out;
    char* ws = (char*)d_ws;
    unsigned short* xh = (unsigned short*)(ws + 0);         // 4 MB
    unsigned short* xl = (unsigned short*)(ws + 4194304);   // 4 MB
    unsigned short* ch = (unsigned short*)(ws + 8388608);   // 4 MB
    unsigned short* cl = (unsigned short*)(ws + 12582912);  // 4 MB
    float* xnorm = (float*)(ws + 16777216);                 // 32 KB
    float* cnorm = (float*)(ws + 16809984);                 // 32 KB
    int*   amin  = (int*)  (ws + 16842752);                 // 32 KB
    int*   counts= (int*)  (ws + 16875520);                 // 32 KB
    float* accum = (float*)(ws + 16908288);                 // 8 B
    float* pmin  = (float*)(ws + 16908544);                 // 256 KB (8 splits x 8192)
    int*   pidx  = (int*)  (ws + 17170688);                 // 256 KB

    vq_init<<<32, 256, 0, stream>>>(counts, accum);
    vq_fill_zero<<<8192, 256, 0, stream>>>((float2*)(out + OFF_ONEHOT), (long long)ONEHOT_LEN / 2);
    vq_prep_x<<<N_TOK, 256, 0, stream>>>(x, xh, xl, xnorm);
    vq_prep_cb<<<K_CB, 256, 0, stream>>>(cb, ch, cl, cnorm);
    dim3 g(64, NSPLIT);
    vq_dist_argmin<<<g, 256, 0, stream>>>(xh, xl, ch, cl, xnorm, cnorm, pmin, pidx);
    vq_amin_combine<<<32, 256, 0, stream>>>(pmin, pidx, amin);
    vq_scatter<<<N_TOK, 256, 0, stream>>>(x, cb, amin, out, counts, accum);
    vq_finalize<<<1, 256, 0, stream>>>(counts, accum, out);
}
